// DecoderBlock_12481174962394
// MI455X (gfx1250) — matrix operand order repaired and checked by reading, not compiled, not run on hardware
//
#include <hip/hip_runtime.h>
#include <hip/hip_bf16.h>

// ---------------------------------------------------------------------------
// CDNA5 (gfx1250) decoder block: LN -> self-MHA -> LN -> cross-MHA -> LN -> FFN
// GEMMs + attention via v_wmma_f32_16x16x32_bf16 (wave32).
// Double-buffered tile staging via GLOBAL_LOAD_ASYNC_TO_LDS_B128 (ASYNCcnt),
// contiguous segments sharing one base + immediate offsets.
// All WMMA B operands pre-transposed so fragment reads are ds_load_b128;
// fragment groups are preloaded before WMMA bursts to hide LDS latency.
// ---------------------------------------------------------------------------

typedef __attribute__((ext_vector_type(16))) __bf16 v16bf;
typedef __attribute__((ext_vector_type(8)))  float  v8f;

__device__ __forceinline__ __bf16 f2bf(float f) {
  unsigned u = __float_as_uint(f);
  u += 0x7FFFu + ((u >> 16) & 1u);          // round-to-nearest-even
  unsigned short h = (unsigned short)(u >> 16);
  __bf16 b;
  __builtin_memcpy(&b, &h, sizeof(h));
  return b;
}

// LDS byte offset of a shared-memory pointer (flat addr low 32 bits = LDS addr).
__device__ __forceinline__ unsigned lds_off(const void* p) {
  return (unsigned)(uintptr_t)p;
}

// Async DMA: 16 bytes/lane global->LDS; INST_OFFSET applies to both sides.
template <int IMM>
__device__ __forceinline__ void async_b128(unsigned lds_addr, const void* gaddr) {
  asm volatile("global_load_async_to_lds_b128 %0, %1, off offset:%2"
               :: "v"(lds_addr), "v"(gaddr), "i"(IMM) : "memory");
}
__device__ __forceinline__ void wait_async0() {
  asm volatile("s_wait_asynccnt 0x0" ::: "memory");
}

// ----------------------------- elementwise ---------------------------------

__global__ void copy_f32(const float* __restrict__ x, float* __restrict__ y, int n) {
  int i = (blockIdx.x * blockDim.x + threadIdx.x) * 4;
  if (i + 3 < n) *(float4*)(y + i) = *(const float4*)(x + i);
}

__global__ void add_f32(float* __restrict__ x, const float* __restrict__ r, int n) {
  int i = (blockIdx.x * blockDim.x + threadIdx.x) * 4;
  if (i + 3 < n) {
    float4 a = *(const float4*)(x + i);
    float4 b = *(const float4*)(r + i);
    a.x += b.x; a.y += b.y; a.z += b.z; a.w += b.w;
    *(float4*)(x + i) = a;
  }
}

// Cast fp32 W[K,N] -> bf16 W^T[N,K] via 32x32 LDS tile.
__global__ __launch_bounds__(256)
void transpose_cast_bf16(const float* __restrict__ W, __bf16* __restrict__ Wt,
                         int K, int N) {
  __shared__ float tile[32][33];
  int n0 = blockIdx.x * 32, k0 = blockIdx.y * 32;
  int tx = threadIdx.x & 31, ty = threadIdx.x >> 5;   // 32 x 8
  #pragma unroll
  for (int i = ty; i < 32; i += 8)
    tile[i][tx] = W[(size_t)(k0 + i) * N + n0 + tx];
  __syncthreads();
  #pragma unroll
  for (int i = ty; i < 32; i += 8)
    Wt[(size_t)(n0 + i) * K + k0 + tx] = f2bf(tile[tx][i]);
}

// Permute bf16 V(b,t,h,d) -> V^T(b,h,d,t) so flash P@V B-frags are K-contiguous.
__global__ __launch_bounds__(256)
void transpose_v_bf16(const __bf16* __restrict__ Vsrc, __bf16* __restrict__ Vt,
                      int Tn, int Hn, int En) {
  __shared__ __bf16 tile[32][33];
  int bh = blockIdx.y >> 1;             // D=64 -> two 32-wide d tiles
  int d0 = (blockIdx.y & 1) * 32;
  int b = bh / Hn, h = bh - b * Hn;
  int t0 = blockIdx.x * 32;
  int tx = threadIdx.x & 31, ty = threadIdx.x >> 5;
  #pragma unroll
  for (int i = ty; i < 32; i += 8)
    tile[i][tx] = Vsrc[(size_t)(b * Tn + t0 + i) * En + h * 64 + d0 + tx];
  __syncthreads();
  #pragma unroll
  for (int i = ty; i < 32; i += 8)
    Vt[((size_t)(b * Hn + h) * 64 + d0 + i) * Tn + t0 + tx] = tile[tx][i];
}

// ------------------------------ layernorm ----------------------------------

__global__ __launch_bounds__(256)
void layernorm_to_bf16(const float* __restrict__ X, const float* __restrict__ G,
                       const float* __restrict__ Bt, __bf16* __restrict__ Y) {
  const int E = 1024;
  int row = blockIdx.x;
  int tid = threadIdx.x;
  const float4 v = ((const float4*)(X + (size_t)row * E))[tid];
  float s  = v.x + v.y + v.z + v.w;
  float ss = v.x * v.x + v.y * v.y + v.z * v.z + v.w * v.w;
  #pragma unroll
  for (int off = 16; off >= 1; off >>= 1) {
    s  += __shfl_down(s, off, 32);
    ss += __shfl_down(ss, off, 32);
  }
  __shared__ float rs[8], rss[8];
  __shared__ float mu_s, rstd_s;
  int w = tid >> 5, ln = tid & 31;
  if (ln == 0) { rs[w] = s; rss[w] = ss; }
  __syncthreads();
  if (tid == 0) {
    float ts = 0.f, tss = 0.f;
    #pragma unroll
    for (int i = 0; i < 8; i++) { ts += rs[i]; tss += rss[i]; }
    float mu  = ts * (1.0f / E);
    float var = tss * (1.0f / E) - mu * mu;
    mu_s = mu; rstd_s = rsqrtf(var + 1e-5f);
  }
  __syncthreads();
  float mu = mu_s, rstd = rstd_s;
  float4 g = ((const float4*)G)[tid];
  float4 b = ((const float4*)Bt)[tid];
  __bf16* yp = Y + (size_t)row * E + tid * 4;
  yp[0] = f2bf((v.x - mu) * rstd * g.x + b.x);
  yp[1] = f2bf((v.y - mu) * rstd * g.y + b.y);
  yp[2] = f2bf((v.z - mu) * rstd * g.z + b.z);
  yp[3] = f2bf((v.w - mu) * rstd * g.w + b.w);
}

// ------------------------------ WMMA GEMM ----------------------------------
// C[M,N_] = A[M,K_](bf16) @ Bt[N_,K_](bf16)^T (+bias, ReLU), out f32 or bf16.
// Block tile 128x64, K-step 64 (8 WMMA/wave/stage), double-buffered async LDS.

template <bool BF16OUT, bool RELU_, bool BIAS_, int N_, int K_>
__global__ __launch_bounds__(256)
void gemm_bf16_wmma(const __bf16* __restrict__ A, const __bf16* __restrict__ Bt,
                    const float* __restrict__ bias, float* __restrict__ Cf,
                    __bf16* __restrict__ Cb) {
  const int BM = 128, BK = 64;
  __shared__ __align__(16) __bf16 As[2][BM * BK];   // 2 x 16 KB, [m][k]
  __shared__ __align__(16) __bf16 Bs[2][64 * BK];   // 2 x  8 KB, [n][k]

  int tid  = threadIdx.x;
  int wave = tid >> 5;
  int lane = tid & 31;
  int lh   = lane >> 4;
  int ll   = lane & 15;
  int m0 = blockIdx.y * BM, n0 = blockIdx.x * 64;

  // Contiguous staging segments (global & LDS strides match -> shared base).
  int arow = tid >> 1, ahalf = (tid & 1) * 32;       // A: half row (64B)
  int brow = tid >> 2, bq    = (tid & 3) * 16;       // B: quarter row (32B)
  const __bf16* gA = &A[(size_t)(m0 + arow) * K_ + ahalf];
  const __bf16* gB = &Bt[(size_t)(n0 + brow) * K_ + bq];
  unsigned lA[2] = { lds_off(&As[0][arow * BK + ahalf]),
                     lds_off(&As[1][arow * BK + ahalf]) };
  unsigned lB[2] = { lds_off(&Bs[0][brow * BK + bq]),
                     lds_off(&Bs[1][brow * BK + bq]) };

  auto stage = [&](int p, int k0) {
    const __bf16* ga = gA + k0;
    const __bf16* gb = gB + k0;
    async_b128<0>(lA[p], ga);  async_b128<16>(lA[p], ga);
    async_b128<32>(lA[p], ga); async_b128<48>(lA[p], ga);
    async_b128<0>(lB[p], gb);  async_b128<16>(lB[p], gb);
  };

  v8f acc[4];
  #pragma unroll
  for (int n = 0; n < 4; n++)
    #pragma unroll
    for (int r = 0; r < 8; r++) acc[n][r] = 0.0f;

  stage(0, 0);
  wait_async0();
  __syncthreads();

  int p = 0;
  for (int k0 = 0; k0 < K_; k0 += BK) {
    if (k0 + BK < K_) stage(p ^ 1, k0 + BK);    // prefetch next tile (async)

    // Preload ALL fragments first (distinct regs) so ds_load_b128s pipeline,
    // then drain the 8 WMMAs back-to-back.
    v16bf af0, af1;
    {
      int ar = wave * 16 + ll;
      #pragma unroll
      for (int j = 0; j < 16; j++) {
        int kk = j + lh * 16;
        af0[j] = As[p][ar * BK + kk];
        af1[j] = As[p][ar * BK + 32 + kk];
      }
    }
    v16bf bf0[4], bf1[4];
    #pragma unroll
    for (int n = 0; n < 4; n++) {
      #pragma unroll
      for (int j = 0; j < 16; j++) {            // contiguous -> ds_load_b128
        int bidx = (n * 16 + ll) * BK + j + lh * 16;
        bf0[n][j] = Bs[p][bidx];
        bf1[n][j] = Bs[p][bidx + 32];
      }
    }
    #pragma unroll
    for (int n = 0; n < 4; n++) {
      acc[n] = __builtin_amdgcn_wmma_f32_16x16x32_bf16(
          false, af0, false, bf0[n], (short)0, acc[n], false, false);
      acc[n] = __builtin_amdgcn_wmma_f32_16x16x32_bf16(
          false, af1, false, bf1[n], (short)0, acc[n], false, false);
    }
    wait_async0();
    __syncthreads();
    p ^= 1;
  }

  #pragma unroll
  for (int n = 0; n < 4; n++) {
    int col = n0 + n * 16 + ll;
    float bv = BIAS_ ? bias[col] : 0.0f;
    #pragma unroll
    for (int r = 0; r < 8; r++) {
      int row = m0 + wave * 16 + lh * 8 + r;
      float v = acc[n][r] + bv;
      if (RELU_) v = fmaxf(v, 0.0f);
      if (BF16OUT) Cb[(size_t)row * N_ + col] = f2bf(v);
      else         Cf[(size_t)row * N_ + col] = v;
    }
  }
}

// --------------------------- flash attention -------------------------------
// One wave per (b,h, 16-row q tile). 32-key chunks: QK^T (4 WMMA), online
// softmax, P@V (4 WMMA). V pre-transposed (b,h,d,t). Double-buffered tiles.

__global__ __launch_bounds__(32)
void flash_attn_wmma(const __bf16* __restrict__ Q, const __bf16* __restrict__ Kb,
                     const __bf16* __restrict__ Vt, float* __restrict__ O,
                     int Tn, int Hn, int En, float scale) {
  const int Dh = 64;
  int lane = threadIdx.x & 31;
  int lh = lane >> 4, ll = lane & 15;
  int qt0 = blockIdx.x * 16;
  int bh = blockIdx.y;
  int b = bh / Hn, h = bh - b * Hn;
  size_t base  = (size_t)b * Tn * En + (size_t)h * Dh;     // q/k rows
  size_t vbase = ((size_t)(b * Hn + h) * Dh) * Tn;         // v^T rows

  __shared__ __align__(16) __bf16 kt[2][32 * 64];   // [token][d]
  __shared__ __align__(16) __bf16 vtT[2][64 * 32];  // [d][token]
  __shared__ __align__(16) __bf16 pt[16 * 32];      // P transpose hop

  // Q fragments for the two 32-wide D chunks (A layout).
  v16bf aq0, aq1;
  {
    size_t qrow = base + (size_t)(qt0 + ll) * En;
    #pragma unroll
    for (int j = 0; j < 16; j++) {
      int d = j + lh * 16;
      aq0[j] = Q[qrow + d];
      aq1[j] = Q[qrow + d + 32];
    }
  }

  // Staging bases: K row per lane (128B), two V^T d-rows per lane (64B each).
  const __bf16* gK0 = Kb + base + (size_t)lane * En;
  const __bf16* gV0 = Vt + vbase + (size_t)(lane * 2) * Tn;
  const __bf16* gV1 = gV0 + Tn;
  unsigned lK[2] = { lds_off(&kt[0][lane * 64]),  lds_off(&kt[1][lane * 64]) };
  unsigned lV[2] = { lds_off(&vtT[0][lane * 64]), lds_off(&vtT[1][lane * 64]) };

  auto stageKV = [&](int p, int c0) {
    const __bf16* gk = gK0 + (size_t)c0 * En;
    async_b128<0>(lK[p], gk);   async_b128<16>(lK[p], gk);
    async_b128<32>(lK[p], gk);  async_b128<48>(lK[p], gk);
    async_b128<64>(lK[p], gk);  async_b128<80>(lK[p], gk);
    async_b128<96>(lK[p], gk);  async_b128<112>(lK[p], gk);
    const __bf16* gv0 = gV0 + c0;
    const __bf16* gv1 = gV1 + c0;
    async_b128<0>(lV[p], gv0);       async_b128<16>(lV[p], gv0);
    async_b128<32>(lV[p], gv0);      async_b128<48>(lV[p], gv0);
    async_b128<0>(lV[p] + 64, gv1);  async_b128<16>(lV[p] + 64, gv1);
    async_b128<32>(lV[p] + 64, gv1); async_b128<48>(lV[p] + 64, gv1);
  };

  v8f accO[4];
  float m[8], l[8];
  #pragma unroll
  for (int r = 0; r < 8; r++) { m[r] = -3.0e38f; l[r] = 0.0f; }
  #pragma unroll
  for (int n = 0; n < 4; n++)
    #pragma unroll
    for (int r = 0; r < 8; r++) accO[n][r] = 0.0f;

  stageKV(0, 0);
  wait_async0();
  __syncthreads();

  int p = 0;
  for (int c0 = 0; c0 < Tn; c0 += 32) {
    if (c0 + 32 < Tn) stageKV(p ^ 1, c0 + 32);   // prefetch next chunk

    // Scores: preload all 4 K-fragments, then 4 WMMAs back-to-back.
    v16bf bk00, bk10, bk01, bk11;   // bk<dchunk><ntile>
    #pragma unroll
    for (int j = 0; j < 16; j++) {
      int d0 = j + lh * 16;
      bk00[j] = kt[p][(ll) * 64 + d0];
      bk10[j] = kt[p][(ll) * 64 + 32 + d0];
      bk01[j] = kt[p][(16 + ll) * 64 + d0];
      bk11[j] = kt[p][(16 + ll) * 64 + 32 + d0];
    }
    v8f s0, s1;
    #pragma unroll
    for (int r = 0; r < 8; r++) { s0[r] = 0.0f; s1[r] = 0.0f; }
    s0 = __builtin_amdgcn_wmma_f32_16x16x32_bf16(false, aq0, false, bk00,
                                                 (short)0, s0, false, false);
    s1 = __builtin_amdgcn_wmma_f32_16x16x32_bf16(false, aq0, false, bk01,
                                                 (short)0, s1, false, false);
    s0 = __builtin_amdgcn_wmma_f32_16x16x32_bf16(false, aq1, false, bk10,
                                                 (short)0, s0, false, false);
    s1 = __builtin_amdgcn_wmma_f32_16x16x32_bf16(false, aq1, false, bk11,
                                                 (short)0, s1, false, false);

    // Online softmax: rows in VGPR slot r (+lh*8), cols across 16 lanes.
    #pragma unroll
    for (int r = 0; r < 8; r++) {
      float a  = s0[r] * scale;
      float bb = s1[r] * scale;
      float mx = fmaxf(a, bb);
      #pragma unroll
      for (int off = 1; off < 16; off <<= 1)
        mx = fmaxf(mx, __shfl_xor(mx, off, 16));
      float mnew = fmaxf(m[r], mx);
      float p0 = __expf(a - mnew), p1 = __expf(bb - mnew);
      float ssum = p0 + p1;
      #pragma unroll
      for (int off = 1; off < 16; off <<= 1)
        ssum += __shfl_xor(ssum, off, 16);
      float corr = __expf(m[r] - mnew);
      m[r] = mnew;
      l[r] = l[r] * corr + ssum;
      #pragma unroll
      for (int n = 0; n < 4; n++) accO[n][r] *= corr;
      int prow = r + lh * 8;                  // C-layout -> LDS transpose hop
      pt[prow * 32 + ll]      = f2bf(p0);
      pt[prow * 32 + 16 + ll] = f2bf(p1);
    }
    __syncthreads();

    // P in A layout; preload all 4 V^T fragments, then 4 WMMAs.
    v16bf ap;
    #pragma unroll
    for (int j = 0; j < 16; j++) {
      int kk = j + lh * 16;
      ap[j] = pt[ll * 32 + kk];
    }
    v16bf bv[4];
    #pragma unroll
    for (int n = 0; n < 4; n++)
      #pragma unroll
      for (int j = 0; j < 16; j++)            // contiguous -> ds_load_b128
        bv[n][j] = vtT[p][(n * 16 + ll) * 32 + j + lh * 16];
    #pragma unroll
    for (int n = 0; n < 4; n++)
      accO[n] = __builtin_amdgcn_wmma_f32_16x16x32_bf16(
          false, ap, false, bv[n], (short)0, accO[n], false, false);

    wait_async0();
    __syncthreads();
    p ^= 1;
  }

  #pragma unroll
  for (int r = 0; r < 8; r++) {
    float inv = 1.0f / l[r];
    size_t orow = base + (size_t)(qt0 + r + lh * 8) * En;
    #pragma unroll
    for (int n = 0; n < 4; n++)
      O[orow + n * 16 + ll] = accO[n][r] * inv;
  }
}

// ------------------------------ launcher -----------------------------------

extern "C" void kernel_launch(void* const* d_in, const int* in_sizes, int n_in,
                              void* d_out, int out_size, void* d_ws, size_t ws_size,
                              hipStream_t stream) {
  (void)in_sizes; (void)n_in; (void)out_size; (void)ws_size;
  const int Bc = 4, Tc = 1024, Ec = 1024, Hc = 16;
  const int Mc = Bc * Tc;      // 4096 tokens
  const int FF = 4 * Ec;       // 4096

  const float* x    = (const float*)d_in[0];
  const float* enc  = (const float*)d_in[1];
  const float* Wq_s = (const float*)d_in[2];
  const float* Wk_s = (const float*)d_in[3];
  const float* Wv_s = (const float*)d_in[4];
  const float* Wq_c = (const float*)d_in[5];
  const float* Wk_c = (const float*)d_in[6];
  const float* Wv_c = (const float*)d_in[7];
  const float* W1   = (const float*)d_in[8];
  const float* b1   = (const float*)d_in[9];
  const float* W2   = (const float*)d_in[10];
  const float* b2   = (const float*)d_in[11];
  const float* ln1_g = (const float*)d_in[12];
  const float* ln1_b = (const float*)d_in[13];
  const float* ln2_g = (const float*)d_in[14];
  const float* ln2_b = (const float*)d_in[15];
  const float* ln3_g = (const float*)d_in[16];
  const float* ln3_b = (const float*)d_in[17];

  char* ws = (char*)d_ws;
  size_t off = 0;
  auto carve = [&](size_t bytes) -> char* {
    char* p = ws + off;
    off += (bytes + 255) & ~(size_t)255;
    return p;
  };
  __bf16* wqs = (__bf16*)carve((size_t)Ec * Ec * 2);
  __bf16* wks = (__bf16*)carve((size_t)Ec * Ec * 2);
  __bf16* wvs = (__bf16*)carve((size_t)Ec * Ec * 2);
  __bf16* wqc = (__bf16*)carve((size_t)Ec * Ec * 2);
  __bf16* wkc = (__bf16*)carve((size_t)Ec * Ec * 2);
  __bf16* wvc = (__bf16*)carve((size_t)Ec * Ec * 2);
  __bf16* w1t = (__bf16*)carve((size_t)Ec * FF * 2);   // (4096 x 1024)
  __bf16* w2t = (__bf16*)carve((size_t)FF * Ec * 2);   // (1024 x 4096)
  __bf16* hq  = (__bf16*)carve((size_t)Mc * Ec * 2);
  __bf16* hk  = (__bf16*)carve((size_t)Mc * Ec * 2);
  __bf16* qb  = (__bf16*)carve((size_t)Mc * Ec * 2);
  __bf16* kb  = (__bf16*)carve((size_t)Mc * Ec * 2);
  __bf16* vb  = (__bf16*)carve((size_t)Mc * Ec * 2);
  __bf16* vtb = (__bf16*)carve((size_t)Mc * Ec * 2);   // V^T (b,h,d,t)
  float*  attn = (float*)carve((size_t)Mc * Ec * 4);
  float*  xcur = (float*)carve((size_t)Mc * Ec * 4);
  __bf16* ffh = (__bf16*)carve((size_t)Mc * FF * 2);

  const int nME = Mc * Ec;
  auto castT = [&](const float* s, __bf16* d, int K, int N) {
    transpose_cast_bf16<<<dim3(N / 32, K / 32), 256, 0, stream>>>(s, d, K, N);
  };
  castT(Wq_s, wqs, Ec, Ec); castT(Wk_s, wks, Ec, Ec); castT(Wv_s, wvs, Ec, Ec);
  castT(Wq_c, wqc, Ec, Ec); castT(Wk_c, wkc, Ec, Ec); castT(Wv_c, wvc, Ec, Ec);
  castT(W1, w1t, Ec, FF);   castT(W2, w2t, FF, Ec);

  copy_f32<<<nME / 1024, 256, 0, stream>>>(x, xcur, nME);

  dim3 gp(Ec / 64, Mc / 128);          // GEMM grid, N=1024
  dim3 gf(FF / 64, Mc / 128);          // GEMM grid, N=4096
  dim3 ga(Tc / 16, Bc * Hc);           // flash grid
  dim3 gv(Tc / 32, 2 * Bc * Hc);       // V-transpose grid
  const float scale = 8.0f;            // sqrt(D) multiply (reference bug)

  // ---- self-attention branch ----
  layernorm_to_bf16<<<Mc, 256, 0, stream>>>(xcur, ln1_g, ln1_b, hq);
  gemm_bf16_wmma<true, false, false, 1024, 1024><<<gp, 256, 0, stream>>>(hq, wqs, nullptr, nullptr, qb);
  gemm_bf16_wmma<true, false, false, 1024, 1024><<<gp, 256, 0, stream>>>(hq, wks, nullptr, nullptr, kb);
  gemm_bf16_wmma<true, false, false, 1024, 1024><<<gp, 256, 0, stream>>>(hq, wvs, nullptr, nullptr, vb);
  transpose_v_bf16<<<gv, 256, 0, stream>>>(vb, vtb, Tc, Hc, Ec);
  flash_attn_wmma<<<ga, 32, 0, stream>>>(qb, kb, vtb, attn, Tc, Hc, Ec, scale);
  add_f32<<<nME / 1024, 256, 0, stream>>>(xcur, attn, nME);

  // ---- cross-attention branch (ln2 on both x and encoder embedding) ----
  layernorm_to_bf16<<<Mc, 256, 0, stream>>>(xcur, ln2_g, ln2_b, hq);
  layernorm_to_bf16<<<Mc, 256, 0, stream>>>(enc, ln2_g, ln2_b, hk);
  gemm_bf16_wmma<true, false, false, 1024, 1024><<<gp, 256, 0, stream>>>(hq, wqc, nullptr, nullptr, qb);
  gemm_bf16_wmma<true, false, false, 1024, 1024><<<gp, 256, 0, stream>>>(hk, wkc, nullptr, nullptr, kb);
  gemm_bf16_wmma<true, false, false, 1024, 1024><<<gp, 256, 0, stream>>>(hk, wvc, nullptr, nullptr, vb);
  transpose_v_bf16<<<gv, 256, 0, stream>>>(vb, vtb, Tc, Hc, Ec);
  flash_attn_wmma<<<ga, 32, 0, stream>>>(qb, kb, vtb, attn, Tc, Hc, Ec, scale);
  add_f32<<<nME / 1024, 256, 0, stream>>>(xcur, attn, nME);

  // ---- FFN branch ----
  layernorm_to_bf16<<<Mc, 256, 0, stream>>>(xcur, ln3_g, ln3_b, hq);
  gemm_bf16_wmma<true, true, true, 4096, 1024><<<gf, 256, 0, stream>>>(hq, w1t, b1, nullptr, ffh);
  gemm_bf16_wmma<false, false, true, 1024, 4096><<<gp, 256, 0, stream>>>(ffh, w2t, b2, attn, nullptr);
  add_f32<<<nME / 1024, 256, 0, stream>>>(xcur, attn, nME);

  // ---- outputs: (x, encoder_embedding) concatenated ----
  float* out = (float*)d_out;
  copy_f32<<<nME / 1024, 256, 0, stream>>>(xcur, out, nME);
  copy_f32<<<nME / 1024, 256, 0, stream>>>(enc, out + (size_t)nME, nME);
}
